// NodeClassifier_59433757442077
// MI455X (gfx1250) — matrix operand order repaired and checked by reading, not compiled, not run on hardware
//
#include <hip/hip_runtime.h>
#include <hip/hip_bf16.h>

typedef __bf16 bf16_t;
typedef __attribute__((ext_vector_type(16))) __bf16 v16bf;
typedef __attribute__((ext_vector_type(8)))  __bf16 v8bf;
typedef __attribute__((ext_vector_type(8)))  float  v8f;

#define ATOMIC_FADD(p, v) __hip_atomic_fetch_add((p), (v), __ATOMIC_RELAXED, __HIP_MEMORY_SCOPE_AGENT)

__device__ __forceinline__ bf16_t f2bf(float f) { return (bf16_t)f; }

// ---------------- degree / norm kernels ----------------

__global__ void k_degrees(const int* __restrict__ src, const int* __restrict__ dst,
                          float* __restrict__ deg_out, float* __restrict__ deg_in, int E) {
  int e = blockIdx.x * blockDim.x + threadIdx.x;
  if (e >= E) return;
  ATOMIC_FADD(&deg_out[src[e]], 1.0f);
  ATOMIC_FADD(&deg_in[dst[e]], 1.0f);
}

__global__ void k_norms(const float* __restrict__ deg_out, const float* __restrict__ deg_in,
                        float* __restrict__ nsrc, float* __restrict__ ndst, int N) {
  int i = blockIdx.x * blockDim.x + threadIdx.x;
  if (i >= N) return;
  float a = deg_out[i], b = deg_in[i];
  nsrc[i] = (a > 0.f) ? rsqrtf(a) : 0.f;
  ndst[i] = (b > 0.f) ? rsqrtf(b) : 0.f;
}

// ---------------- bf16 split-prep kernels ----------------

// A1 = x * norm_src, split into bf16 hi/lo  (row-major N x H)
__global__ void k_prep_x(const float* __restrict__ x, const float* __restrict__ nsrc,
                         bf16_t* __restrict__ Ah, bf16_t* __restrict__ Al, int N, int H) {
  int idx = blockIdx.x * blockDim.x + threadIdx.x;
  if (idx >= N * H) return;
  int i = idx / H;
  float v = x[idx] * nsrc[i];
  bf16_t h = f2bf(v);
  Ah[idx] = h;
  Al[idx] = f2bf(v - (float)h);
}

// Transpose weight [K x Nout] row-major -> column-major [NoutPad x K], zero-pad cols, split hi/lo
__global__ void k_prep_w(const float* __restrict__ W, bf16_t* __restrict__ Wth,
                         bf16_t* __restrict__ Wtl, int K, int Nout, int NoutPad) {
  int idx = blockIdx.x * blockDim.x + threadIdx.x;
  if (idx >= NoutPad * K) return;
  int n = idx / K;
  int k = idx % K;
  float v = (n < Nout) ? W[k * Nout + n] : 0.f;
  bf16_t h = f2bf(v);
  Wth[idx] = h;
  Wtl[idx] = f2bf(v - (float)h);
}

// ---------------- WMMA GEMM: C[M x ncols] = A[M x K] * Bt[colsPad x K]^T ----------------
// bf16x3 error-compensated: Ahi*Bhi + Alo*Bhi + Ahi*Blo  (~fp32 accuracy)
// One 16x16 tile per wave32. M must be a multiple of 16, K a multiple of 32.

__global__ void __launch_bounds__(256)
k_wmma_gemm(const bf16_t* __restrict__ Ah, const bf16_t* __restrict__ Al,
            const bf16_t* __restrict__ Bth, const bf16_t* __restrict__ Btl,
            float* __restrict__ C, int M, int K, int ncols, int ldc, int tiles_n) {
  int waveId = (blockIdx.x * blockDim.x + threadIdx.x) >> 5;
  int lane   = threadIdx.x & 31;
  int tiles_m = M >> 4;
  if (waveId >= tiles_m * tiles_n) return;   // uniform per wave: EXEC stays all-1s for WMMA
  int tm = waveId / tiles_n;
  int tn = waveId % tiles_n;

  int half = lane >> 4;       // 0: lanes 0-15, 1: lanes 16-31
  int l15  = lane & 15;
  const bf16_t* arh = Ah + (size_t)(tm * 16 + l15) * K;
  const bf16_t* arl = Al + (size_t)(tm * 16 + l15) * K;
  int bcol = tn * 16 + l15;
  const bf16_t* bch = Bth + (size_t)bcol * K;
  const bf16_t* bcl = Btl + (size_t)bcol * K;

  v8f acc = {};
  for (int k0 = 0; k0 < K; k0 += 32) {
    // A 16-bit 16x32 layout: lane half sel picks K runs {0..7,16..23} or {8..15,24..31}
    v8bf a0h = *(const v8bf*)(arh + k0 + half * 8);
    v8bf a1h = *(const v8bf*)(arh + k0 + 16 + half * 8);
    v8bf a0l = *(const v8bf*)(arl + k0 + half * 8);
    v8bf a1l = *(const v8bf*)(arl + k0 + 16 + half * 8);
    // B 32x16: same K runs as A: lanes 0-15 hold K {0..7,16..23}, lanes 16-31 hold {8..15,24..31} (column-major source)
    v8bf b0h = *(const v8bf*)(bch + k0 + half * 8);
    v8bf b1h = *(const v8bf*)(bch + k0 + 16 + half * 8);
    v8bf b0l = *(const v8bf*)(bcl + k0 + half * 8);
    v8bf b1l = *(const v8bf*)(bcl + k0 + 16 + half * 8);

    v16bf vah, val, vbh, vbl;
#pragma unroll
    for (int i = 0; i < 8; ++i) {
      vah[i] = a0h[i]; vah[8 + i] = a1h[i];
      val[i] = a0l[i]; val[8 + i] = a1l[i];
      vbh[i] = b0h[i]; vbh[8 + i] = b1h[i];
      vbl[i] = b0l[i]; vbl[8 + i] = b1l[i];
    }
    acc = __builtin_amdgcn_wmma_f32_16x16x32_bf16(false, vah, false, vbh, (short)0, acc, false, false);
    acc = __builtin_amdgcn_wmma_f32_16x16x32_bf16(false, val, false, vbh, (short)0, acc, false, false);
    acc = __builtin_amdgcn_wmma_f32_16x16x32_bf16(false, vah, false, vbl, (short)0, acc, false, false);
  }

  if (bcol < ncols) {
#pragma unroll
    for (int v = 0; v < 8; ++v) {
      int crow = tm * 16 + v + half * 8;   // C/D layout: M = vgpr + 8*(lane>=16), N = lane&15
      C[(size_t)crow * ldc + bcol] = acc[v];
    }
  }
}

// ---------------- edge scatter-add: agg[dst] += t[src], F features ----------------

__global__ void k_scatter(const float* __restrict__ t, float* __restrict__ agg,
                          const int* __restrict__ src, const int* __restrict__ dst,
                          long long total, int F) {
  long long idx = (long long)blockIdx.x * blockDim.x + threadIdx.x;
  if (idx >= total) return;
  int e = (int)(idx / F);
  int f = (int)(idx % F);
  int s = src[e], d = dst[e];
  ATOMIC_FADD(&agg[(size_t)d * F + f], t[(size_t)s * F + f]);
}

// ---------------- post layer 1: relu(agg*ndst + b1) * nsrc -> bf16 hi/lo ----------------

__global__ void k_post1(const float* __restrict__ agg1, const float* __restrict__ ndst,
                        const float* __restrict__ nsrc, const float* __restrict__ b1,
                        bf16_t* __restrict__ A2h, bf16_t* __restrict__ A2l, int N, int F) {
  int idx = blockIdx.x * blockDim.x + threadIdx.x;
  if (idx >= N * F) return;
  int i = idx / F;
  int f = idx % F;
  float v = agg1[idx] * ndst[i] + b1[f];
  v = v > 0.f ? v : 0.f;            // relu
  float g = v * nsrc[i];            // pre-apply layer-2 source norm
  bf16_t h = f2bf(g);
  A2h[idx] = h;
  A2l[idx] = f2bf(g - (float)h);
}

// ---------------- post layer 2: out = relu(agg*ndst + b2) ----------------

__global__ void k_post2(const float* __restrict__ agg2, const float* __restrict__ ndst,
                        const float* __restrict__ b2, float* __restrict__ out, int N, int F) {
  int idx = blockIdx.x * blockDim.x + threadIdx.x;
  if (idx >= N * F) return;
  int i = idx / F;
  int f = idx % F;
  float v = agg2[idx] * ndst[i] + b2[f];
  out[idx] = v > 0.f ? v : 0.f;
}

// ---------------- host launcher ----------------

extern "C" void kernel_launch(void* const* d_in, const int* in_sizes, int n_in,
                              void* d_out, int out_size, void* d_ws, size_t ws_size,
                              hipStream_t stream) {
  const float* x   = (const float*)d_in[0];
  const int*   src = (const int*)d_in[1];
  const int*   dst = (const int*)d_in[2];
  const float* W1  = (const float*)d_in[3];
  const float* b1  = (const float*)d_in[4];
  const float* W2  = (const float*)d_in[5];
  const float* b2  = (const float*)d_in[6];
  (void)n_in; (void)out_size; (void)ws_size;

  const int E  = in_sizes[1];           // 1,600,000 edges
  const int H2 = in_sizes[4];           // 64 (b1 length)
  const int H1 = in_sizes[3] / H2;      // 128
  const int N  = in_sizes[0] / H1;      // 100,000 nodes (multiple of 16)
  const int C  = in_sizes[6];           // 40 classes
  const int CP = (C + 15) & ~15;        // 48 (padded WMMA tile columns)

  // workspace carve-out (256B aligned)
  size_t off = 0;
  auto alloc = [&](size_t bytes) -> void* {
    size_t o = (off + 255) & ~(size_t)255;
    off = o + bytes;
    return (void*)((char*)d_ws + o);
  };
  float*  deg_out = (float*)alloc((size_t)N * 4);
  float*  deg_in  = (float*)alloc((size_t)N * 4);
  float*  nsrc    = (float*)alloc((size_t)N * 4);
  float*  ndst    = (float*)alloc((size_t)N * 4);
  bf16_t* A1h = (bf16_t*)alloc((size_t)N * H1 * 2);
  bf16_t* A1l = (bf16_t*)alloc((size_t)N * H1 * 2);
  bf16_t* W1h = (bf16_t*)alloc((size_t)H2 * H1 * 2);
  bf16_t* W1l = (bf16_t*)alloc((size_t)H2 * H1 * 2);
  float*  t1   = (float*)alloc((size_t)N * H2 * 4);
  float*  agg1 = (float*)alloc((size_t)N * H2 * 4);
  bf16_t* A2h = (bf16_t*)alloc((size_t)N * H2 * 2);
  bf16_t* A2l = (bf16_t*)alloc((size_t)N * H2 * 2);
  bf16_t* W2h = (bf16_t*)alloc((size_t)CP * H2 * 2);
  bf16_t* W2l = (bf16_t*)alloc((size_t)CP * H2 * 2);
  float*  t2   = (float*)alloc((size_t)N * C * 4);
  float*  agg2 = (float*)alloc((size_t)N * C * 4);

  // zero accumulators (graph-capturable memset nodes)
  hipMemsetAsync(deg_out, 0, (size_t)N * 4, stream);
  hipMemsetAsync(deg_in,  0, (size_t)N * 4, stream);
  hipMemsetAsync(agg1,    0, (size_t)N * H2 * 4, stream);
  hipMemsetAsync(agg2,    0, (size_t)N * C * 4, stream);

  const int BT = 256;
  k_degrees<<<(E + BT - 1) / BT, BT, 0, stream>>>(src, dst, deg_out, deg_in, E);
  k_norms<<<(N + BT - 1) / BT, BT, 0, stream>>>(deg_out, deg_in, nsrc, ndst, N);
  k_prep_x<<<(int)(((size_t)N * H1 + BT - 1) / BT), BT, 0, stream>>>(x, nsrc, A1h, A1l, N, H1);
  k_prep_w<<<(H2 * H1 + BT - 1) / BT, BT, 0, stream>>>(W1, W1h, W1l, H1, H2, H2);
  k_prep_w<<<(CP * H2 + BT - 1) / BT, BT, 0, stream>>>(W2, W2h, W2l, H2, C, CP);

  // layer 1: t1 = (x*nsrc) @ W1   [N x 64], K=128
  {
    int tiles_n = H2 / 16;                       // 4
    int waves = (N / 16) * tiles_n;              // 25000
    k_wmma_gemm<<<(waves * 32 + BT - 1) / BT, BT, 0, stream>>>(
        A1h, A1l, W1h, W1l, t1, N, H1, H2, H2, tiles_n);
  }
  k_scatter<<<(int)(((size_t)E * H2 + BT - 1) / BT), BT, 0, stream>>>(
      t1, agg1, src, dst, (long long)E * H2, H2);
  k_post1<<<(int)(((size_t)N * H2 + BT - 1) / BT), BT, 0, stream>>>(
      agg1, ndst, nsrc, b1, A2h, A2l, N, H2);

  // layer 2: t2 = (h1*nsrc) @ W2   [N x 40], K=64, tiles padded to 48 cols, store masked
  {
    int tiles_n = CP / 16;                       // 3
    int waves = (N / 16) * tiles_n;              // 18750
    k_wmma_gemm<<<(waves * 32 + BT - 1) / BT, BT, 0, stream>>>(
        A2h, A2l, W2h, W2l, t2, N, H2, C, C, tiles_n);
  }
  k_scatter<<<(int)(((size_t)E * C + BT - 1) / BT), BT, 0, stream>>>(
      t2, agg2, src, dst, (long long)E * C, C);
  k_post2<<<(int)(((size_t)N * C + BT - 1) / BT), BT, 0, stream>>>(
      agg2, ndst, b2, (float*)d_out, N, C);
}
